// HetConv_14276471292308
// MI455X (gfx1250) — matrix operand order repaired and checked by reading, not compiled, not run on hardware
//
#include <hip/hip_runtime.h>
#include <hip/hip_bf16.h>

// ---------------- problem constants ----------------
#define N_IMG 32
#define C_IN  256
#define C_OUT 256
#define H_IMG 64
#define W_IMG 64
#define KTOT  768          // 256 center + 8 taps * 64 sparse channels
#define SA_STRIDE 264      // padded row stride (elements) -> bank-conflict-free frags
#define SH_XS 68           // padded halo x slots (66 used)

typedef __attribute__((ext_vector_type(16))) __bf16 v16bf;
typedef __attribute__((ext_vector_type(8)))  __bf16 v8bf;
typedef __attribute__((ext_vector_type(8)))  float  v8f;
typedef __attribute__((ext_vector_type(4)))  unsigned int u32x4;
typedef __attribute__((ext_vector_type(8)))  int i32x8;
typedef __attribute__((ext_vector_type(4)))  int i32x4;

// =====================================================================
// Prep: pack effective weights once into bf16, layout [r][j][k], o=4j+r
//   k in [0,256): center tap; W = ((i&3)==r) ? Wk[o][i][1][1] : W1[o][i]
//   k in [256,768): k-256 = tap*64+cc, i = 4cc+r, tap skips center
// =====================================================================
__global__ __launch_bounds__(256) void pack_weights(const float* __restrict__ Wk,
                                                    const float* __restrict__ W1,
                                                    __bf16* __restrict__ wpack) {
#if defined(__AMDGCN__)
    int idx = blockIdx.x * 256 + threadIdx.x;           // 4*64*768 = 196608 exact
    int k = idx % KTOT;
    int j = (idx / KTOT) & 63;
    int r = idx / (KTOT * 64);
    int o = 4 * j + r;
    float v;
    if (k < 256) {
        int i = k;
        v = ((i & 3) == r) ? Wk[((o * C_IN + i) * 3 + 1) * 3 + 1]
                           : W1[o * C_IN + i];
    } else {
        int ks  = k - 256;
        int tap = ks >> 6;
        int cc  = ks & 63;
        int i9  = tap + (tap >= 4 ? 1 : 0);             // skip center (idx 4)
        int ky = i9 / 3, kx = i9 % 3;
        int i  = 4 * cc + r;
        v = Wk[((o * C_IN + i) * 3 + ky) * 3 + kx];
    }
    wpack[idx] = (__bf16)v;
#endif
}

// =====================================================================
// Main: one block per (n, y). 64 spatial x 256 outputs, K=768 via WMMA.
// =====================================================================
__global__ __launch_bounds__(256) void hetconv_wmma(const float* __restrict__ x,
                                                    const __bf16* __restrict__ wpack,
                                                    float* __restrict__ out) {
#if defined(__AMDGCN__)
    __shared__ __align__(16) float  sRaw[C_IN * W_IMG];            // 64 KB: TDM target / C-buffer
    __shared__ __align__(32) __bf16 sA[W_IMG * SA_STRIDE];         // 33 KB: transposed acts [x][c]
    __shared__ __align__(32) __bf16 sH[3 * SH_XS * SA_STRIDE];     // 105 KB: halo [ky][xslot][c'] (residue-major)

    const int tid = threadIdx.x;
    const int s   = blockIdx.x;
    const int n   = s >> 6;
    const int y   = s & 63;
    const float* xbase = x + (size_t)n * C_IN * H_IMG * W_IMG + (size_t)y * W_IMG;

    // ---- TDM: DMA x[n, 0:256, y, 0:64] (256 rows x 256B, stride 16KB) into sRaw
    if (tid < 32) {
        unsigned long long ga = (unsigned long long)(uintptr_t)(const void*)xbase;
        unsigned lds_addr = (unsigned)(uintptr_t)(void*)&sRaw[0];
        u32x4 g0;
        g0.x = 1u;                                          // count=1, no gather
        g0.y = lds_addr;                                    // lds_addr [63:32]
        g0.z = (unsigned)(ga & 0xFFFFFFFFull);              // global_addr lo
        g0.w = (unsigned)((ga >> 32) & 0x01FFFFFFull) | (2u << 30); // addr hi | type=2
        i32x8 g1;
        g1[0] = (int)(2u << 16);          // data_size = 4B, no mask/pad/iterate
        g1[1] = (int)(64u << 16);         // tensor_dim0 = 64   (bits 79:48)
        g1[2] = (int)(256u << 16);        // tensor_dim1 = 256  (bits 111:80)
        g1[3] = (int)(64u << 16);         // tile_dim0 = 64     (bits 127:112)
        g1[4] = 256;                      // tile_dim1 = 256
        g1[5] = 4096;                     // tensor_dim0_stride = H*W elems
        g1[6] = 0;
        g1[7] = 0;
        i32x4 z4 = {0, 0, 0, 0};
#if defined(__clang_major__) && (__clang_major__ >= 23)
        i32x8 z8 = {0, 0, 0, 0, 0, 0, 0, 0};
        __builtin_amdgcn_tensor_load_to_lds(g0, g1, z4, z4, z8, 0);
#else
        __builtin_amdgcn_tensor_load_to_lds(g0, g1, z4, z4, 0);
#endif
    }

    // ---- stage 3-row halo (overlaps TDM). Channel order residue-major:
    //      c' = (c&3)*64 + (c>>2)  => sparse frags are contiguous per residue.
    for (int task = tid; task < 3 * 66 * 4; task += 256) {
        int rr  = task / (66 * 4);
        int rem = task % (66 * 4);
        int xs  = rem >> 2;              // 0..65 -> input x = xs-1
        int cg  = rem & 3;               // residue group (64 channels)
        int iy = y + rr - 1;
        int ix = xs - 1;
        bool valid = (iy >= 0) && (iy < H_IMG) && (ix >= 0) && (ix < W_IMG);
        __bf16 tmp[64];
        #pragma unroll
        for (int cl = 0; cl < 64; ++cl) {
            int c = 4 * cl + cg;
            float v = valid ? x[(((size_t)n * C_IN + c) * H_IMG + iy) * W_IMG + ix] : 0.0f;
            tmp[cl] = (__bf16)v;
        }
        __bf16* dst = &sH[(rr * SH_XS + xs) * SA_STRIDE + cg * 64];
        #pragma unroll
        for (int q = 0; q < 8; ++q)
            *(v8bf*)&dst[q * 8] = *(v8bf*)&tmp[q * 8];
    }

    if (tid < 32) __builtin_amdgcn_s_wait_tensorcnt(0);
    __syncthreads();

    // ---- in-LDS transpose + fp32->bf16: sRaw[c][x] -> sA[x][c]
    {
        int g = tid >> 6, p = tid & 63;
        __bf16 tmp[64];
        #pragma unroll
        for (int cl = 0; cl < 64; ++cl)
            tmp[cl] = (__bf16)sRaw[(g * 64 + cl) * W_IMG + p];
        __bf16* dst = &sA[p * SA_STRIDE + g * 64];
        #pragma unroll
        for (int q = 0; q < 8; ++q)
            *(v8bf*)&dst[q * 8] = *(v8bf*)&tmp[q * 8];
    }
    __syncthreads();

    // ---- WMMA main loop: M=spatial, N=outputs (residue-grouped), K=768
    const int wv   = tid >> 5;
    const int lane = tid & 31;
    const int half = lane >> 4;          // K-half select (this code's contiguous k = 16*half + s)
    const int l16  = lane & 15;
    const int rw   = wv >> 1;            // residue of this wave's output columns
    const int jjm0 = ((wv & 1) * 32) + l16;           // j within residue group
    const __bf16* wrow0 = wpack + (size_t)(rw * 64 + jjm0) * KTOT;
    const __bf16* wrow1 = wrow0 + (size_t)16 * KTOT;
    __builtin_prefetch(wrow0, 0, 1);
    __builtin_prefetch(wrow1, 0, 1);

    const v8f vzero = {0.f, 0.f, 0.f, 0.f, 0.f, 0.f, 0.f, 0.f};
    v8f acc[4][2];
    #pragma unroll
    for (int mT = 0; mT < 4; ++mT) { acc[mT][0] = vzero; acc[mT][1] = vzero; }

    // center phase: K = 0..255 over all input channels, row y, no shift
    #pragma unroll
    for (int ks = 0; ks < 8; ++ks) {
        v16bf b0 = *(const v16bf*)(wrow0 + ks * 32 + half * 16);
        v16bf b1 = *(const v16bf*)(wrow1 + ks * 32 + half * 16);
        #pragma unroll
        for (int mT = 0; mT < 4; ++mT) {
            const __bf16* ar = &sA[(mT * 16 + l16) * SA_STRIDE + ks * 32];
            v8bf lo = *(const v8bf*)(ar + half * 16);
            v8bf hi = *(const v8bf*)(ar + half * 16 + 8);
            v16bf a = __builtin_shufflevector(lo, hi, 0,1,2,3,4,5,6,7,8,9,10,11,12,13,14,15);
            acc[mT][0] = __builtin_amdgcn_wmma_f32_16x16x32_bf16(false, a, false, b0, (short)0, acc[mT][0], false, false);
            acc[mT][1] = __builtin_amdgcn_wmma_f32_16x16x32_bf16(false, a, false, b1, (short)0, acc[mT][1], false, false);
        }
    }

    // sparse phase: 8 off-center taps x 64 channels (i = 4cc + rw) from halo
    #pragma unroll
    for (int st = 0; st < 16; ++st) {
        const int tap = st >> 1;
        const int i9  = tap + (tap >= 4 ? 1 : 0);
        const int ky  = i9 / 3, kx = i9 % 3;
        const int cb  = (st & 1) * 32;
        v16bf b0 = *(const v16bf*)(wrow0 + 256 + st * 32 + half * 16);
        v16bf b1 = *(const v16bf*)(wrow1 + 256 + st * 32 + half * 16);
        #pragma unroll
        for (int mT = 0; mT < 4; ++mT) {
            const int p = mT * 16 + l16;
            const __bf16* ar = &sH[(ky * SH_XS + (p + kx)) * SA_STRIDE + rw * 64 + cb];
            v8bf lo = *(const v8bf*)(ar + half * 16);
            v8bf hi = *(const v8bf*)(ar + half * 16 + 8);
            v16bf a = __builtin_shufflevector(lo, hi, 0,1,2,3,4,5,6,7,8,9,10,11,12,13,14,15);
            acc[mT][0] = __builtin_amdgcn_wmma_f32_16x16x32_bf16(false, a, false, b0, (short)0, acc[mT][0], false, false);
            acc[mT][1] = __builtin_amdgcn_wmma_f32_16x16x32_bf16(false, a, false, b1, (short)0, acc[mT][1], false, false);
        }
    }

    // ---- epilogue: stage C through LDS (sRaw reused), then coalesced stores
    {
        float* cbuf = sRaw;                       // 256 cols x 64 spatial = 64 KB
        const int jj0 = rw * 64 + jjm0;           // logical column (residue-grouped)
        #pragma unroll
        for (int mT = 0; mT < 4; ++mT) {
            #pragma unroll
            for (int v = 0; v < 8; ++v) {
                int m = mT * 16 + v + 8 * half;   // spatial row (C-matrix layout)
                cbuf[jj0 * 64 + m]        = acc[mT][0][v];
                cbuf[(jj0 + 16) * 64 + m] = acc[mT][1][v];
            }
        }
    }
    __syncthreads();
    #pragma unroll
    for (int it = 0; it < 4; ++it) {
        int task = tid + 256 * it;                // 256 out rows x 4 segments
        int orow = task >> 2, seg = task & 3;
        int jj = (orow & 3) * 64 + (orow >> 2);   // inverse of o = 4*(jj&63)+(jj>>6)
        const float4* src = (const float4*)&sRaw[jj * 64 + seg * 16];
        float4* dst = (float4*)(out + (((size_t)n * C_OUT + orow) * H_IMG + y) * W_IMG + seg * 16);
        dst[0] = src[0]; dst[1] = src[1]; dst[2] = src[2]; dst[3] = src[3];
    }
#endif
}

extern "C" void kernel_launch(void* const* d_in, const int* in_sizes, int n_in,
                              void* d_out, int out_size, void* d_ws, size_t ws_size,
                              hipStream_t stream) {
    const float* x  = (const float*)d_in[0];
    const float* Wk = (const float*)d_in[1];
    const float* W1 = (const float*)d_in[2];
    float* out = (float*)d_out;
    __bf16* wpack = (__bf16*)d_ws;               // 4*64*768 bf16 = 384 KB scratch

    pack_weights<<<768, 256, 0, stream>>>(Wk, W1, wpack);      // 196608 elems
    hetconv_wmma<<<N_IMG * H_IMG, 256, 0, stream>>>(x, wpack, out);
    (void)in_sizes; (void)n_in; (void)out_size; (void)ws_size;
}
